// GaussianKernel_738734375297
// MI455X (gfx1250) — matrix operand order repaired and checked by reading, not compiled, not run on hardware
//
#include <hip/hip_runtime.h>
#include <hip/hip_bf16.h>

// KAN Gaussian-RBF layer on gfx1250 (wave32):
//   out[16384,512] = exp(-((x[...,None]-grid)/h)^2).reshape(B,4096) @ W[4096,512]
//
// Tiered strategy on ws_size:
//  (1) >=136MB: basis materialized ONCE as bf16 in d_ws (kills the 4x redundant
//      exp work), then a pure bf16 WMMA GEMM with BOTH tiles staged via
//      global_load_async_to_lds_b128 (ASYNCcnt). HBM ~324MB => ~14us floor.
//  (2) >=4MB: fused basis-in-registers + async-staged prepacked bf16 W.
//  (3) else : fully fused, W converted in-loop.
// All paths: v_wmma_f32_16x16x32_bf16 core, ping-pong LDS, 1 barrier/K-tile.

typedef __attribute__((ext_vector_type(16))) __bf16 v16bf;
typedef __attribute__((ext_vector_type(8)))  __bf16 v8bf;
typedef __attribute__((ext_vector_type(8)))  float  v8f;
typedef __attribute__((ext_vector_type(2)))  __bf16 v2bf;

#define IN_F  512
#define OUT_F 512
#define NG    8
#define KDIM  (IN_F * NG)   // 4096
#define BM    128
#define BN    128
#define BK    64
#define FPI   8
#define LDA   72            // padded LDS stride (bf16 elems), 144B rows
#define LDB   72
#define NITER (KDIM / BK)   // 64

#if __has_builtin(__builtin_amdgcn_exp2f)
#define EXP2F __builtin_amdgcn_exp2f
#else
#define EXP2F exp2f
#endif

static __device__ inline unsigned pk2bf(float lo, float hi) {
#if __has_builtin(__builtin_amdgcn_cvt_pk_bf16_f32)
    v2bf r = __builtin_amdgcn_cvt_pk_bf16_f32(lo, hi);
    return __builtin_bit_cast(unsigned, r);
#else
    unsigned ul = __builtin_bit_cast(unsigned, lo);
    unsigned uh = __builtin_bit_cast(unsigned, hi);
    ul += 0x7FFFu + ((ul >> 16) & 1u);
    uh += 0x7FFFu + ((uh >> 16) & 1u);
    return (ul >> 16) | (uh & 0xFFFF0000u);
#endif
}

static __device__ inline void async_copy_b128(unsigned lds_off, const void* gptr) {
    asm volatile("global_load_async_to_lds_b128 %0, %1, off"
                 :: "v"(lds_off), "v"(gptr) : "memory");
}
static __device__ inline void wait_async0() {
#if __has_builtin(__builtin_amdgcn_s_wait_asynccnt)
    __builtin_amdgcn_s_wait_asynccnt(0);
#else
    asm volatile("s_wait_asynccnt 0x0" ::: "memory");
#endif
}

static __device__ inline void load_grid(const float* grid, float* gr) {
    const float4 g0 = ((const float4*)grid)[0];
    const float4 g1 = ((const float4*)grid)[1];
    gr[0] = g0.x; gr[1] = g0.y; gr[2] = g0.z; gr[3] = g0.w;
    gr[4] = g1.x; gr[5] = g1.y; gr[6] = g1.z; gr[7] = g1.w;
}

#define CEXP (-3.0625f * 1.44269504088896340736f) // -(1/h)^2*log2e, h=4/7

// ---------------- one-time: W[4096][512] f32 -> bf16 N-major [512][4096]
__global__ __launch_bounds__(256) void prepack_w(const float* __restrict__ w,
                                                 unsigned* __restrict__ wpk) {
    const int k2 = blockIdx.x;
    const int k  = k2 * 2;
    for (int n = threadIdx.x; n < OUT_F; n += 256) {
        const float a = w[(size_t)k * OUT_F + n];
        const float b = w[(size_t)(k + 1) * OUT_F + n];
        wpk[(size_t)n * (KDIM / 2) + k2] = pk2bf(a, b);
    }
}

// ---------------- pass 1: basis(x) -> bf16 A_ws[BATCH][4096], computed once
__global__ __launch_bounds__(256) void basis_pass(
    const float* __restrict__ x, const float* __restrict__ grid,
    unsigned short* __restrict__ apk) {
    const int idx = blockIdx.x * 256 + threadIdx.x;  // one per 4 features
    const int row = idx >> 7;                        // 128 groups per row
    const int fg  = idx & 127;
    float gr[8];
    load_grid(grid, gr);
    const float4 xv4 = *(const float4*)&x[(size_t)row * IN_F + fg * 4];
    const float xv[4] = { xv4.x, xv4.y, xv4.z, xv4.w };
    #pragma unroll
    for (int j = 0; j < 4; ++j) {
        float e[NG];
        #pragma unroll
        for (int g = 0; g < NG; ++g) {
            const float d = xv[j] - gr[g];
            e[g] = EXP2F(CEXP * d * d);
        }
        uint4 q;
        q.x = pk2bf(e[0], e[1]); q.y = pk2bf(e[2], e[3]);
        q.z = pk2bf(e[4], e[5]); q.w = pk2bf(e[6], e[7]);
        *(uint4*)&apk[(size_t)row * KDIM + (fg * 4 + j) * NG] = q;
    }
}

// ---------------- shared epilogue
static __device__ inline void store_acc(v8f (&acc)[2][4], float* __restrict__ out,
                                        int m0, int n0, int wm, int wn,
                                        int half, int l16) {
    #pragma unroll
    for (int mi = 0; mi < 2; ++mi)
        #pragma unroll
        for (int ni = 0; ni < 4; ++ni) {
            const int col = n0 + wn * 64 + ni * 16 + l16;
            #pragma unroll
            for (int r = 0; r < 8; ++r) {
                const int row = m0 + wm * 32 + mi * 16 + half * 8 + r;
                __builtin_nontemporal_store(acc[mi][ni][r],
                                            &out[(size_t)row * OUT_F + col]);
            }
        }
}

// ---------------- shared WMMA tile compute
static __device__ inline void wmma_tile(const __bf16* __restrict__ As,
                                        const __bf16* __restrict__ Bs,
                                        v8f (&acc)[2][4],
                                        int wm, int wn, int half, int l16) {
    #pragma unroll
    for (int ks = 0; ks < 2; ++ks) {
        const int kk = ks * 32;
        v16bf a[2], b[4];
        #pragma unroll
        for (int mi = 0; mi < 2; ++mi) {
            const __bf16* p = &As[(wm * 32 + mi * 16 + l16) * LDA + kk + half * 8];
            const v8bf lo = *(const v8bf*)p;
            const v8bf hi = *(const v8bf*)(p + 16);
            a[mi] = __builtin_shufflevector(lo, hi,
                    0,1,2,3,4,5,6,7,8,9,10,11,12,13,14,15);
        }
        #pragma unroll
        for (int ni = 0; ni < 4; ++ni) {
            const __bf16* p = &Bs[(wn * 64 + ni * 16 + l16) * LDB + kk + half * 8];
            const v8bf lo = *(const v8bf*)p;
            const v8bf hi = *(const v8bf*)(p + 16);
            b[ni] = __builtin_shufflevector(lo, hi,
                    0,1,2,3,4,5,6,7,8,9,10,11,12,13,14,15);
        }
        #pragma unroll
        for (int mi = 0; mi < 2; ++mi)
            #pragma unroll
            for (int ni = 0; ni < 4; ++ni)
                acc[mi][ni] = __builtin_amdgcn_wmma_f32_16x16x32_bf16(
                    false, a[mi], false, b[ni], (short)0, acc[mi][ni], false, false);
    }
}

// ---------------- pass 2: pure bf16 GEMM, A and B async-staged from d_ws
__global__ __launch_bounds__(256) void kan_gemm_ws(
    const unsigned short* __restrict__ apk,   // bf16 [BATCH][4096]
    const unsigned short* __restrict__ bpk,   // bf16 [512][4096] (N-major)
    float* __restrict__ out)
{
    __shared__ __align__(16) __bf16 As[2][BM * LDA];
    __shared__ __align__(16) __bf16 Bs[2][BN * LDB];

    const int tid  = threadIdx.x;
    const int lane = tid & 31;
    const int wave = tid >> 5;
    const int wm = wave & 3, wn = wave >> 2;
    const int half = lane >> 4, l16 = lane & 15;
    const int n0 = blockIdx.x * BN;
    const int m0 = blockIdx.y * BM;

    v8f acc[2][4];
    {
        v8f z;
        #pragma unroll
        for (int e = 0; e < 8; ++e) z[e] = 0.0f;
        #pragma unroll
        for (int i = 0; i < 2; ++i)
            #pragma unroll
            for (int j = 0; j < 4; ++j) acc[i][j] = z;
    }

    auto stage = [&](int it, int buf) {
        const int k0 = it * BK;
        #pragma unroll
        for (int i = 0; i < 4; ++i) {       // A: 128 rows x 8 16B chunks
            const int cidx = tid + 256 * i;
            const int row = cidx >> 3, ck = cidx & 7;
            async_copy_b128((unsigned)(size_t)&As[buf][row * LDA + ck * 8],
                            &apk[(size_t)(m0 + row) * KDIM + k0 + ck * 8]);
        }
        #pragma unroll
        for (int i = 0; i < 4; ++i) {       // B: 128 rows x 8 16B chunks
            const int cidx = tid + 256 * i;
            const int row = cidx >> 3, ck = cidx & 7;
            async_copy_b128((unsigned)(size_t)&Bs[buf][row * LDB + ck * 8],
                            &bpk[(size_t)(n0 + row) * KDIM + k0 + ck * 8]);
        }
    };

    stage(0, 0);
    wait_async0();
    __syncthreads();
    for (int it = 0; it < NITER; ++it) {
        const int buf = it & 1;
        if (it + 1 < NITER) stage(it + 1, buf ^ 1);
        wmma_tile(As[buf], Bs[buf], acc, wm, wn, half, l16);
        wait_async0();
        __syncthreads();
    }
    store_acc(acc, out, m0, n0, wm, wn, half, l16);
}

// ---------------- fused kernel (fallback tiers)
template <bool PRE>
__global__ __launch_bounds__(256) void kan_rbf_fused(
    const float* __restrict__ x, const float* __restrict__ grid,
    const float* __restrict__ w, const unsigned short* __restrict__ wpk,
    float* __restrict__ out)
{
    __shared__ __align__(16) __bf16 As[2][BM * LDA];
    __shared__ __align__(16) __bf16 Bs[2][BN * LDB];

    const int tid  = threadIdx.x;
    const int lane = tid & 31;
    const int wave = tid >> 5;
    const int wm = wave & 3, wn = wave >> 2;
    const int half = lane >> 4, l16 = lane & 15;
    const int n0 = blockIdx.x * BN;
    const int m0 = blockIdx.y * BM;

    float gr[8];
    load_grid(grid, gr);

    v8f acc[2][4];
    {
        v8f z;
        #pragma unroll
        for (int e = 0; e < 8; ++e) z[e] = 0.0f;
        #pragma unroll
        for (int i = 0; i < 2; ++i)
            #pragma unroll
            for (int j = 0; j < 4; ++j) acc[i][j] = z;
    }

    const int xrow = tid >> 1;
    const int fgrp = tid & 1;
    const int wc4  = tid & 31;
    const int wr2  = (tid >> 5) * 2;

    auto stageA = [&](int it, int buf) {
        const int f0 = it * FPI;
        const float4 xv4 = *(const float4*)&x[(size_t)(m0 + xrow) * IN_F + f0 + fgrp * 4];
        const float xv[4] = { xv4.x, xv4.y, xv4.z, xv4.w };
        #pragma unroll
        for (int j = 0; j < 4; ++j) {
            float e[NG];
            #pragma unroll
            for (int g = 0; g < NG; ++g) {
                const float d = xv[j] - gr[g];
                e[g] = EXP2F(CEXP * d * d);
            }
            uint4 q;
            q.x = pk2bf(e[0], e[1]); q.y = pk2bf(e[2], e[3]);
            q.z = pk2bf(e[4], e[5]); q.w = pk2bf(e[6], e[7]);
            *(uint4*)&As[buf][xrow * LDA + (fgrp * 4 + j) * NG] = q;
        }
    };

    auto stageB = [&](int it, int buf) {
        const int k0 = it * BK;
        if (PRE) {
            #pragma unroll
            for (int i = 0; i < 4; ++i) {
                const int cidx = tid + 256 * i;
                const int row = cidx >> 3, ck = cidx & 7;
                async_copy_b128((unsigned)(size_t)&Bs[buf][row * LDB + ck * 8],
                                &wpk[(size_t)(n0 + row) * KDIM + k0 + ck * 8]);
            }
        } else {
            #pragma unroll
            for (int rr = 0; rr < 4; ++rr) {
                const int r = rr * 16 + wr2;
                const float4 w0 = *(const float4*)&w[(size_t)(k0 + r)     * OUT_F + n0 + wc4 * 4];
                const float4 w1 = *(const float4*)&w[(size_t)(k0 + r + 1) * OUT_F + n0 + wc4 * 4];
                *(unsigned*)&Bs[buf][(wc4 * 4 + 0) * LDB + r] = pk2bf(w0.x, w1.x);
                *(unsigned*)&Bs[buf][(wc4 * 4 + 1) * LDB + r] = pk2bf(w0.y, w1.y);
                *(unsigned*)&Bs[buf][(wc4 * 4 + 2) * LDB + r] = pk2bf(w0.z, w1.z);
                *(unsigned*)&Bs[buf][(wc4 * 4 + 3) * LDB + r] = pk2bf(w0.w, w1.w);
            }
        }
    };

    stageA(0, 0);
    stageB(0, 0);
    if (PRE) wait_async0();
    __syncthreads();
    for (int it = 0; it < NITER; ++it) {
        const int buf = it & 1;
        if (it + 1 < NITER) { stageA(it + 1, buf ^ 1); stageB(it + 1, buf ^ 1); }
        wmma_tile(As[buf], Bs[buf], acc, wm, wn, half, l16);
        if (PRE) wait_async0();
        __syncthreads();
    }
    store_acc(acc, out, m0, n0, wm, wn, half, l16);
}

extern "C" void kernel_launch(void* const* d_in, const int* in_sizes, int n_in,
                              void* d_out, int out_size, void* d_ws, size_t ws_size,
                              hipStream_t stream) {
    const float* x    = (const float*)d_in[0];
    const float* grid = (const float*)d_in[1];
    const float* w    = (const float*)d_in[2];
    float*       out  = (float*)d_out;

    const int batch = in_sizes[0] / IN_F;     // 16384
    dim3 g(OUT_F / BN, batch / BM);           // (4, 128)

    const size_t wOff  = 0;
    const size_t wB    = (size_t)KDIM * OUT_F * 2;          // 4 MB
    const size_t aOff  = 8u << 20;                          // A at +8MB
    const size_t aB    = (size_t)batch * KDIM * 2;          // 128 MB

    if (ws_size >= aOff + aB) {
        // two-pass: basis once + prepacked W + pure async WMMA GEMM
        unsigned short* bpk = (unsigned short*)((char*)d_ws + wOff);
        unsigned short* apk = (unsigned short*)((char*)d_ws + aOff);
        prepack_w<<<KDIM / 2, 256, 0, stream>>>(w, (unsigned*)bpk);
        basis_pass<<<(batch * (IN_F / 4)) / 256, 256, 0, stream>>>(x, grid, apk);
        kan_gemm_ws<<<g, 256, 0, stream>>>(apk, bpk, out);
    } else if (ws_size >= wB) {
        unsigned short* bpk = (unsigned short*)d_ws;
        prepack_w<<<KDIM / 2, 256, 0, stream>>>(w, (unsigned*)bpk);
        kan_rbf_fused<true><<<g, 256, 0, stream>>>(x, grid, w, bpk, out);
    } else {
        kan_rbf_fused<false><<<g, 256, 0, stream>>>(x, grid, w, nullptr, out);
    }
}
